// HeatKernelLoss_22342419874460
// MI455X (gfx1250) — compile-verified, matrix operand order checked by reading, not run on hardware
//
#include <hip/hip_runtime.h>
#include <math.h>

// ---------------------------------------------------------------------------
// HeatKernelLoss on MI455X (gfx1250), wave32 + WMMA bf16.
//
// Strategy (bandwidth-bound workload):
//   * Convert L_pred / L_target to bf16 once  -> all repeated matvec passes
//     read half the bytes; both bf16 matrices (151 MB) fit in the 192 MB L2.
//   * All matvecs are a skinny GEMM  Y(16 x n) = (A x X^T) using
//     v_wmma_f32_16x16x32_bf16, f32 accumulate.  The 16-row "X" block is
//     transposed (contiguous along K) and staged per-block into LDS
//     (conflict-free padded layout), A is streamed with b128 loads +
//     global_prefetch_b8.  Two independent accumulators per wave break the
//     WMMA RAW chain (2 wmma / loop body).
//   * Split-K=8 with partial buffers + deterministic reduction (no atomics).
//   * Power iteration (20 its) -> lambda; Lanczos (m=10, 8 probes batched);
//     16-lane Jacobi eigensolver for the 10x10 tridiagonals; final MSE.
//
// Workspace requirement: 2*n^2*2 bytes + ~6 MB  (~157 MB for n=6144).
// ---------------------------------------------------------------------------

typedef __bf16 bf16_t;
typedef __attribute__((ext_vector_type(16))) __bf16 v16bf;
typedef __attribute__((ext_vector_type(8)))  __bf16 v8bf;
typedef __attribute__((ext_vector_type(4)))  __bf16 v4bf;
typedef __attribute__((ext_vector_type(8)))  float  v8f;

#define KSPLIT   8
#define NPROBES  8
#define MDEG     10
#define NCOLS    16   // padded column block (8 probes / 1 power vector)
#define TPB      8    // row tiles per block (= waves per 256-thread block)
#define LDSPAD   8    // bf16 elements of row padding in LDS B-slice

__device__ __forceinline__ unsigned pcg_hash(unsigned x) {
  x = x * 747796405u + 2891336453u;
  x = ((x >> ((x >> 28) + 4u)) ^ x) * 277803737u;
  return (x >> 22) ^ x;
}

// ---------------- f32 -> bf16 matrix convert (streaming, vectorized) -------
__global__ __launch_bounds__(256)
void k_cvt_bf16(const float* __restrict__ src, bf16_t* __restrict__ dst, size_t n4) {
  size_t stride = (size_t)gridDim.x * blockDim.x;
  for (size_t i = (size_t)blockIdx.x * blockDim.x + threadIdx.x; i < n4; i += stride) {
    float4 v = ((const float4*)src)[i];
    v4bf o;
    o.x = (bf16_t)v.x; o.y = (bf16_t)v.y; o.z = (bf16_t)v.z; o.w = (bf16_t)v.w;
    ((v4bf*)dst)[i] = o;
  }
}

// ---------------- zero floats ----------------------------------------------
__global__ __launch_bounds__(256)
void k_zero(float* __restrict__ p, size_t cnt) {
  size_t stride = (size_t)gridDim.x * blockDim.x;
  for (size_t i = (size_t)blockIdx.x * blockDim.x + threadIdx.x; i < cnt; i += stride)
    p[i] = 0.0f;
}

__global__ void k_init_one(float* p) {
  if (threadIdx.x == 0 && blockIdx.x == 0) p[0] = 1.0f;
}

// ---------------- RNG: power-iteration start vector (row 0 of Yt) ----------
__global__ __launch_bounds__(256)
void k_rand_uniform(float* __restrict__ v, int n, unsigned seed) {
  int stride = gridDim.x * blockDim.x;
  for (int i = blockIdx.x * blockDim.x + threadIdx.x; i < n; i += stride) {
    unsigned h = pcg_hash(seed ^ (unsigned)(i * 0x9E3779B9u));
    v[i] = ((float)(h >> 8) * (1.0f / 16777216.0f)) * 2.0f - 1.0f;
  }
}

// ---------------- RNG: Rademacher probes /sqrt(n); fills V (f32) and Xt ----
__global__ __launch_bounds__(256)
void k_probes(float* __restrict__ V, bf16_t* __restrict__ Xt, int n, unsigned seed) {
  float inv = 1.0f / sqrtf((float)n);
  size_t tot = (size_t)NCOLS * n;
  size_t stride = (size_t)gridDim.x * blockDim.x;
  for (size_t i = (size_t)blockIdx.x * blockDim.x + threadIdx.x; i < tot; i += stride) {
    int p = (int)(i / n);
    int j = (int)(i % n);
    float val = 0.0f;
    if (p < NPROBES) {
      unsigned h = pcg_hash(seed ^ (unsigned)(p * 0x85EBCA6Bu) ^ (unsigned)(j * 0xC2B2AE35u));
      val = (h & 1u) ? inv : -inv;
    }
    V[i]  = val;
    Xt[i] = (bf16_t)val;
  }
}

// ---------------- core WMMA skinny GEMM ------------------------------------
// Ypart[slice][col][row] = sum_{k in slice} A[row][k] * Xt[col][k]
// A : n x n bf16 row-major.  Xt : 16 x n bf16 (column block, transposed).
// Block = 8 waves sharing ONE K slice across 8 row tiles; the 16 x klen
// B-slice is staged once into LDS (padded, bank-conflict-free), A streams
// from global (L2-hot).  Two accumulators -> 2 independent WMMAs per iter.
__global__ __launch_bounds__(256)
void k_gemm_wmma(const bf16_t* __restrict__ A, const bf16_t* __restrict__ Xt,
                 float* __restrict__ Ypart, int n) {
  extern __shared__ bf16_t Bs[];     // [16][klen + LDSPAD]

  int klen  = n / KSPLIT;
  int klenp = klen + LDSPAD;         // 776 for n=6144: 16B-aligned rows,
                                     // dword stride 388 % 64 == 4 -> 0 conflicts
  int slice  = blockIdx.x % KSPLIT;
  int tgroup = blockIdx.x / KSPLIT;
  int kbeg   = slice * klen;

  // cooperative copy of B-slice into LDS: 16 rows x klen bf16, b128 chunks
  int chunks = (16 * klen) >> 3;     // v8bf chunks
  int cpr    = klen >> 3;            // chunks per row
  for (int c = threadIdx.x; c < chunks; c += blockDim.x) {
    int row = c / cpr;
    int o8  = (c - row * cpr) << 3;
    *(v8bf*)&Bs[row * klenp + o8] = *(const v8bf*)(Xt + (size_t)row * n + kbeg + o8);
  }
  __syncthreads();

  int wave = threadIdx.x >> 5;
  int lane = threadIdx.x & 31;
  int tile = tgroup * TPB + wave;
  int r = lane & 15;                 // A row in tile / B column / D column N
  int h = lane >> 4;                 // lane half selects K sub-group (ISA layout)

  const bf16_t* Arow = A  + (size_t)(tile * 16 + r) * n + kbeg;
  const bf16_t* Bl   = Bs + r * klenp;

  v8f acc0 = {}, acc1 = {};
  for (int kk = 0; kk < klen; kk += 64) {
    __builtin_prefetch(Arow + kk + 1024, 0, 1);            // global_prefetch_b8
    // A fragments (16-bit 16x32 layout): half h covers K=8h..+7 and 16+8h..+7
    v8bf a0lo = *(const v8bf*)(Arow + kk + 8 * h);
    v8bf a0hi = *(const v8bf*)(Arow + kk + 16 + 8 * h);
    v8bf a1lo = *(const v8bf*)(Arow + kk + 32 + 8 * h);
    v8bf a1hi = *(const v8bf*)(Arow + kk + 48 + 8 * h);
    // B fragments from LDS (ds_load_b128, conflict-free)
    v8bf b0lo = *(const v8bf*)(Bl + kk + 8 * h);
    v8bf b0hi = *(const v8bf*)(Bl + kk + 16 + 8 * h);
    v8bf b1lo = *(const v8bf*)(Bl + kk + 32 + 8 * h);
    v8bf b1hi = *(const v8bf*)(Bl + kk + 48 + 8 * h);
    v16bf av0 = __builtin_shufflevector(a0lo, a0hi, 0,1,2,3,4,5,6,7,8,9,10,11,12,13,14,15);
    v16bf bv0 = __builtin_shufflevector(b0lo, b0hi, 0,1,2,3,4,5,6,7,8,9,10,11,12,13,14,15);
    v16bf av1 = __builtin_shufflevector(a1lo, a1hi, 0,1,2,3,4,5,6,7,8,9,10,11,12,13,14,15);
    v16bf bv1 = __builtin_shufflevector(b1lo, b1hi, 0,1,2,3,4,5,6,7,8,9,10,11,12,13,14,15);
    acc0 = __builtin_amdgcn_wmma_f32_16x16x32_bf16(false, av0, false, bv0,
                                                   (short)0, acc0, false, false);
    acc1 = __builtin_amdgcn_wmma_f32_16x16x32_bf16(false, av1, false, bv1,
                                                   (short)0, acc1, false, false);
  }
  v8f acc = acc0 + acc1;

  // D layout: lane -> N = lane%16, VGPR d -> M = d + 8*(lane/16)
  float* Yp = Ypart + (size_t)slice * NCOLS * n;
  int rowbase = tile * 16 + 8 * h;
#pragma unroll
  for (int d = 0; d < 8; ++d)
    Yp[(size_t)r * n + rowbase + d] = acc[d];
}

// ---------------- split-K reduce (+ scale by *sptr) ------------------------
__global__ __launch_bounds__(256)
void k_reduce_scale(float* __restrict__ Yt, const float* __restrict__ Ypart,
                    int n, const float* __restrict__ sptr) {
  size_t tot = (size_t)NCOLS * n;
  float  s   = *sptr;
  size_t stride = (size_t)gridDim.x * blockDim.x;
  for (size_t i = (size_t)blockIdx.x * blockDim.x + threadIdx.x; i < tot; i += stride) {
    float acc = 0.0f;
#pragma unroll
    for (int sl = 0; sl < KSPLIT; ++sl) acc += Ypart[(size_t)sl * tot + i];
    Yt[i] = acc * s;
  }
}

// ---------------- per-row dot products (deterministic, 1 block / row) ------
__global__ __launch_bounds__(256)
void k_dots(const float* __restrict__ a, const float* __restrict__ b,
            float* __restrict__ out, int n) {
  __shared__ float red[256];
  int row = blockIdx.x;
  const float* ar = a + (size_t)row * n;
  const float* br = b + (size_t)row * n;
  float s = 0.0f;
  for (int i = threadIdx.x; i < n; i += blockDim.x) s += ar[i] * br[i];
  red[threadIdx.x] = s;
  __syncthreads();
  for (int st = blockDim.x >> 1; st > 0; st >>= 1) {
    if ((int)threadIdx.x < st) red[threadIdx.x] += red[threadIdx.x + st];
    __syncthreads();
  }
  if (threadIdx.x == 0) out[row] = red[0];
}

// ---------------- normalize rows by 1/(sqrt(ssq)+eps); encode to Xt --------
__global__ __launch_bounds__(256)
void k_normenc(const float* __restrict__ src, float* __restrict__ fout,
               bf16_t* __restrict__ Xt, const float* __restrict__ ssq,
               int n, int rows, float eps) {
  size_t tot = (size_t)rows * n;
  size_t stride = (size_t)gridDim.x * blockDim.x;
  for (size_t i = (size_t)blockIdx.x * blockDim.x + threadIdx.x; i < tot; i += stride) {
    int p = (int)(i / n);
    float inv = 1.0f / (sqrtf(ssq[p]) + eps);
    float v = src[i] * inv;
    if (fout) fout[i] = v;
    Xt[i] = (bf16_t)v;
  }
}

// ---------------- w[p][i] = y[p][i] - c_p * v[p][i]  (c = cptr or sqrt) ----
__global__ __launch_bounds__(256)
void k_wcomb(float* __restrict__ w, const float* __restrict__ y,
             const float* __restrict__ v, const float* __restrict__ cptr,
             int n, int rows, int csqrt) {
  size_t tot = (size_t)rows * n;
  size_t stride = (size_t)gridDim.x * blockDim.x;
  for (size_t i = (size_t)blockIdx.x * blockDim.x + threadIdx.x; i < tot; i += stride) {
    int p = (int)(i / n);
    float c = cptr[p];
    if (csqrt) c = sqrtf(c);
    w[i] = y[i] - c * v[i];
  }
}

// ---------------- lambda finalize: lam=|dot|, scl=1/(lam+1e-8) -------------
__global__ void k_lamfin(const float* __restrict__ dot, float* lam, float* scl) {
  if (threadIdx.x == 0 && blockIdx.x == 0) {
    float l = fabsf(dot[0]);
    lam[0] = l;
    scl[0] = 1.0f / (l + 1e-8f);
  }
}

// ---------------- 10x10 tridiag eigensolve (Jacobi) + heat weights ---------
// 16 independent problems (2 matrices x 8 probes), one per lane.
__global__ void k_eig_heat(const float* __restrict__ ALPHA,  // [2][10][16]
                           const float* __restrict__ BETSQ,  // [2][10][16]
                           const float* __restrict__ LAM,    // [2]
                           float* __restrict__ PP) {         // [2][2][8]
  int t = threadIdx.x;
  if (t >= 16) return;
  int m = t >> 3, p = t & 7;

  float A[MDEG][MDEG];
  float vrow[MDEG];
#pragma unroll
  for (int i = 0; i < MDEG; ++i) {
#pragma unroll
    for (int j = 0; j < MDEG; ++j) A[i][j] = 0.0f;
    vrow[i] = (i == 0) ? 1.0f : 0.0f;
  }
  for (int j = 0; j < MDEG; ++j)
    A[j][j] = ALPHA[(m * MDEG + j) * 16 + p];
  for (int j = 1; j < MDEG; ++j) {
    float b = sqrtf(BETSQ[(m * MDEG + j) * 16 + p]);
    A[j - 1][j] = b;
    A[j][j - 1] = b;
  }

  for (int sweep = 0; sweep < 14; ++sweep) {
    for (int pi = 0; pi < MDEG - 1; ++pi) {
      for (int q = pi + 1; q < MDEG; ++q) {
        float apq = A[pi][q];
        if (fabsf(apq) > 1e-20f) {
          float tau = (A[q][q] - A[pi][pi]) / (2.0f * apq);
          float tt  = copysignf(1.0f, tau) / (fabsf(tau) + sqrtf(1.0f + tau * tau));
          float c   = 1.0f / sqrtf(1.0f + tt * tt);
          float s   = tt * c;
          for (int k = 0; k < MDEG; ++k) {          // columns: B = A*J
            float akp = A[k][pi], akq = A[k][q];
            A[k][pi] = c * akp - s * akq;
            A[k][q]  = s * akp + c * akq;
          }
          for (int k = 0; k < MDEG; ++k) {          // rows: A' = J^T*B
            float apk = A[pi][k], aqk = A[q][k];
            A[pi][k] = c * apk - s * aqk;
            A[q][k]  = s * apk + c * aqk;
          }
          float vp = vrow[pi], vq = vrow[q];
          vrow[pi] = c * vp - s * vq;
          vrow[q]  = s * vp + c * vq;
        }
      }
    }
  }

  float lam = LAM[m];
  const float tlist[2] = {0.5f, 1.0f};
  for (int ti = 0; ti < 2; ++ti) {
    float ts = tlist[ti] * lam;
    float sum = 0.0f;
    for (int k = 0; k < MDEG; ++k)
      sum += vrow[k] * vrow[k] * expf(-ts * A[k][k]);
    PP[(m * 2 + ti) * NPROBES + p] = sum;
  }
}

// ---------------- final MSE -------------------------------------------------
__global__ void k_final(const float* __restrict__ PP, float* __restrict__ out) {
  if (threadIdx.x == 0 && blockIdx.x == 0) {
    float loss = 0.0f;
    for (int ti = 0; ti < 2; ++ti) {
      float tp = 0.0f, tt = 0.0f;
      for (int p = 0; p < NPROBES; ++p) {
        tp += PP[(0 * 2 + ti) * NPROBES + p];
        tt += PP[(1 * 2 + ti) * NPROBES + p];
      }
      tp *= (1.0f / NPROBES);
      tt *= (1.0f / NPROBES);
      float d = tp - tt;
      loss += d * d;
    }
    out[0] = loss * 0.5f;
  }
}

// ===========================================================================
extern "C" void kernel_launch(void* const* d_in, const int* in_sizes, int n_in,
                              void* d_out, int out_size, void* d_ws, size_t ws_size,
                              hipStream_t stream) {
  const float* Lsrc[2] = {(const float*)d_in[0], (const float*)d_in[1]};
  int n = (int)(sqrt((double)in_sizes[0]) + 0.5);   // 6144
  size_t nsq  = (size_t)n * n;
  size_t vecN = (size_t)NCOLS * n;                  // 16*n elements

  // ---------------- workspace layout (needs ~2*n^2*2 + 6MB bytes) ---------
  char* w = (char*)d_ws;
  bf16_t* Abf[2];
  Abf[0] = (bf16_t*)w;            w += nsq * sizeof(bf16_t);
  Abf[1] = (bf16_t*)w;            w += nsq * sizeof(bf16_t);
  bf16_t* Xt    = (bf16_t*)w;     w += vecN * sizeof(bf16_t);
  float*  Ypart = (float*)w;      w += (size_t)KSPLIT * vecN * sizeof(float);
  float*  Yt    = (float*)w;      w += vecN * sizeof(float);
  float*  Vold  = (float*)w;      w += vecN * sizeof(float);
  float*  Vnew  = (float*)w;      w += vecN * sizeof(float);
  float*  Wb    = (float*)w;      w += vecN * sizeof(float);
  float*  SS    = (float*)w;      w += 16 * sizeof(float);
  float*  DOT   = (float*)w;      w += 16 * sizeof(float);
  float*  LAM   = (float*)w;      w += 2 * sizeof(float);
  float*  SCL   = (float*)w;      w += 2 * sizeof(float);
  float*  ONE   = (float*)w;      w += 4 * sizeof(float);
  float*  ALPHA = (float*)w;      w += 2 * MDEG * 16 * sizeof(float);
  float*  BETSQ = (float*)w;      w += 2 * MDEG * 16 * sizeof(float);
  float*  PP    = (float*)w;      w += 2 * 2 * NPROBES * sizeof(float);
  (void)ws_size; (void)n_in; (void)out_size;

  // ---------------- launch geometry ---------------------------------------
  int ntiles     = n / 16;                                  // 384
  int klen       = n / KSPLIT;                              // 768
  int gemmBlocks = (ntiles / TPB) * KSPLIT;                 // 384
  int gemmShmem  = 16 * (klen + LDSPAD) * (int)sizeof(bf16_t);  // ~24.3 KB
  int vecBlocks  = (int)((vecN + 255) / 256);               // 384
  int rowBlocks  = (int)((n + 255) / 256);                  // 24
  int cvtBlocks  = 4096;

  k_init_one<<<1, 32, 0, stream>>>(ONE);

  for (int m = 0; m < 2; ++m) {
    unsigned seedPow   = 0x1234u + 7919u * (unsigned)m;
    unsigned seedProbe = 0xBEEFu + 104729u * (unsigned)m;

    // bf16 copy of L (single streaming pass; subsequent passes are L2-hot)
    k_cvt_bf16<<<cvtBlocks, 256, 0, stream>>>(Lsrc[m], Abf[m], nsq / 4);

    // zero full Xt (power phase only touches row 0)
    k_zero<<<vecBlocks, 256, 0, stream>>>((float*)Xt, vecN / 2);

    // ---- power iteration: lam = |v . (L v)| -----------------------------
    k_rand_uniform<<<rowBlocks, 256, 0, stream>>>(Yt, n, seedPow);
    for (int it = 0; it < 20; ++it) {
      k_dots<<<1, 256, 0, stream>>>(Yt, Yt, SS, n);
      k_normenc<<<rowBlocks, 256, 0, stream>>>(Yt, (float*)nullptr, Xt, SS, n, 1, 1e-12f);
      k_gemm_wmma<<<gemmBlocks, 256, gemmShmem, stream>>>(Abf[m], Xt, Ypart, n);
      k_reduce_scale<<<vecBlocks, 256, 0, stream>>>(Yt, Ypart, n, ONE);
    }
    k_dots<<<1, 256, 0, stream>>>(Yt, Yt, SS, n);
    k_normenc<<<rowBlocks, 256, 0, stream>>>(Yt, Wb, Xt, SS, n, 1, 1e-12f);
    k_gemm_wmma<<<gemmBlocks, 256, gemmShmem, stream>>>(Abf[m], Xt, Ypart, n);
    k_reduce_scale<<<vecBlocks, 256, 0, stream>>>(Yt, Ypart, n, ONE);
    k_dots<<<1, 256, 0, stream>>>(Wb, Yt, DOT, n);
    k_lamfin<<<1, 32, 0, stream>>>(DOT, LAM + m, SCL + m);

    // ---- Lanczos, 8 probes batched --------------------------------------
    float* aBase = ALPHA + m * MDEG * 16;
    float* bBase = BETSQ + m * MDEG * 16;

    k_probes<<<vecBlocks, 256, 0, stream>>>(Vold, Xt, n, seedProbe);
    k_gemm_wmma<<<gemmBlocks, 256, gemmShmem, stream>>>(Abf[m], Xt, Ypart, n);
    k_reduce_scale<<<vecBlocks, 256, 0, stream>>>(Yt, Ypart, n, SCL + m);
    k_dots<<<NPROBES, 256, 0, stream>>>(Vold, Yt, aBase, n);            // a0
    k_wcomb<<<vecBlocks, 256, 0, stream>>>(Wb, Yt, Vold, aBase, n, NPROBES, 0);

    float* vo = Vold;
    float* vn = Vnew;
    for (int j = 1; j < MDEG; ++j) {
      float* aj = aBase + j * 16;
      float* bj = bBase + j * 16;
      k_dots<<<NPROBES, 256, 0, stream>>>(Wb, Wb, bj, n);               // b_j^2
      k_normenc<<<vecBlocks, 256, 0, stream>>>(Wb, vn, Xt, bj, n, NPROBES, 1e-30f);
      k_gemm_wmma<<<gemmBlocks, 256, gemmShmem, stream>>>(Abf[m], Xt, Ypart, n);
      k_reduce_scale<<<vecBlocks, 256, 0, stream>>>(Yt, Ypart, n, SCL + m);
      k_wcomb<<<vecBlocks, 256, 0, stream>>>(Wb, Yt, vo, bj, n, NPROBES, 1); // w = y - b*v_old
      k_dots<<<NPROBES, 256, 0, stream>>>(vn, Wb, aj, n);               // a_j
      k_wcomb<<<vecBlocks, 256, 0, stream>>>(Wb, Wb, vn, aj, n, NPROBES, 0); // w -= a*v_new
      float* tmp = vo; vo = vn; vn = tmp;
    }
  }

  // ---- 10x10 eigensolve + heat weights, then MSE --------------------------
  k_eig_heat<<<1, 32, 0, stream>>>(ALPHA, BETSQ, LAM, PP);
  k_final<<<1, 32, 0, stream>>>(PP, (float*)d_out);
}
